// ReusableAttention_63230508532545
// MI455X (gfx1250) — matrix operand order repaired and checked by reading, not compiled, not run on hardware
//
#include <hip/hip_runtime.h>
#include <hip/hip_bf16.h>

#define B_  2
#define S_  2048
#define D_  1024
#define H_  16
#define HD_ 64

typedef __bf16 bf16_t;
typedef __attribute__((ext_vector_type(16))) __bf16 v16bf;
typedef __attribute__((ext_vector_type(8)))  __bf16 v8bf;
typedef __attribute__((ext_vector_type(8)))  float  v8f;
typedef int v4i __attribute__((ext_vector_type(4)));

#if __has_builtin(__builtin_amdgcn_global_load_async_to_lds_b128)
#define HAS_ASYNC_LDS 1
#endif

typedef __attribute__((address_space(1))) v4i* gptr_v4i;
typedef __attribute__((address_space(3))) v4i* lptr_v4i;

// 16-byte global -> LDS copy. Async (ASYNCcnt) on gfx1250 toolchains that
// expose the builtin; falls back to a synchronous vector copy otherwise.
static __device__ __forceinline__ void cp16(bf16_t* lds_dst, const bf16_t* gsrc) {
#ifdef HAS_ASYNC_LDS
  __builtin_amdgcn_global_load_async_to_lds_b128(
      (gptr_v4i)(void*)gsrc, (lptr_v4i)(void*)lds_dst, 0, 0);
#else
  *(uint4*)lds_dst = *(const uint4*)gsrc;
#endif
}

static __device__ __forceinline__ void wait_async0() {
#ifdef HAS_ASYNC_LDS
#if __has_builtin(__builtin_amdgcn_s_wait_asynccnt)
  __builtin_amdgcn_s_wait_asynccnt(0);
#else
  asm volatile("s_wait_asynccnt 0" ::: "memory");
#endif
#endif
}

static __device__ __forceinline__ bf16_t to_bf16(float f) {
  unsigned u = __builtin_bit_cast(unsigned, f);
  u = (u + 0x7FFFu + ((u >> 16) & 1u)) >> 16;          // RNE to bf16
  unsigned short s = (unsigned short)u;
  return __builtin_bit_cast(bf16_t, s);
}

// A-matrix 16x32 bf16 fragment from row-major (K-contiguous) tile.
// ISA 7.12.2: lanes 0-15 -> row=lane, K 0..7 & 16..23; lanes 16-31 -> K 8..15 & 24..31.
static __device__ __forceinline__ v16bf frag_a(const bf16_t* tile, int stride, int lane) {
  int m = lane & 15;
  const bf16_t* p = tile + m * stride + ((lane & 16) ? 8 : 0);
  v8bf lo = *(const v8bf*)p;
  v8bf hi = *(const v8bf*)(p + 16);
  v16bf r;
#pragma unroll
  for (int i = 0; i < 8; ++i) { r[i] = lo[i]; r[i + 8] = hi[i]; }
  return r;
}

// B-matrix 32x16 bf16 fragment from [N][K] (K-contiguous) tile.
static __device__ __forceinline__ v16bf frag_b(const bf16_t* tile, int stride, int lane) {
  int n = lane & 15;
  const bf16_t* p = tile + n * stride + ((lane & 16) ? 8 : 0);
  v8bf lo = *(const v8bf*)p;
  v8bf hi = *(const v8bf*)(p + 16);
  v16bf r;
#pragma unroll
  for (int i = 0; i < 8; ++i) { r[i] = lo[i]; r[i + 8] = hi[i]; }
  return r;
}

static __device__ __forceinline__ v8f wmma_bf16(v16bf a, v16bf b, v8f c) {
  return __builtin_amdgcn_wmma_f32_16x16x32_bf16(false, a, false, b, (short)0, c, false, false);
}

// ---------------- conversion kernels ----------------
__global__ __launch_bounds__(256) void cvt_x_kernel(const float* __restrict__ x,
                                                    bf16_t* __restrict__ o, int n) {
  int i = blockIdx.x * 256 + threadIdx.x;
  if (i < n) o[i] = to_bf16(x[i]);
}

// in: fp32 [K=1024][N=1024] -> out: bf16 [N][K]
__global__ __launch_bounds__(256) void cvt_wT_kernel(const float* __restrict__ in,
                                                     bf16_t* __restrict__ out) {
  __shared__ float tile[32][33];
  int bx = blockIdx.x * 32, by = blockIdx.y * 32;
  int tx = threadIdx.x & 31, ty = threadIdx.x >> 5;
  for (int i = ty; i < 32; i += 8) tile[i][tx] = in[(size_t)(bx + i) * D_ + by + tx];
  __syncthreads();
  for (int i = ty; i < 32; i += 8) out[(size_t)(by + i) * D_ + bx + tx] = to_bf16(tile[tx][i]);
}

// ---------------- QKV projection GEMM (double-buffered async pipeline) ----------------
__global__ __launch_bounds__(256) void gemm_qkv_kernel(
    const bf16_t* __restrict__ A, const bf16_t* __restrict__ Wq,
    const bf16_t* __restrict__ Wk, const bf16_t* __restrict__ Wv,
    const float* __restrict__ bq, const float* __restrict__ bk, const float* __restrict__ bv,
    bf16_t* __restrict__ Qo, bf16_t* __restrict__ Ko, bf16_t* __restrict__ Vo) {
  __shared__ __align__(16) bf16_t As[2][128 * 40];
  __shared__ __align__(16) bf16_t Bs[2][64 * 40];
  const int tid = threadIdx.x, lane = tid & 31, wave = tid >> 5;
  const int wm = wave & 3, wn = wave >> 2;
  const int m0 = blockIdx.x * 128, n0 = blockIdx.y * 64;
  const int z = blockIdx.z;
  const bf16_t* W = (z == 0) ? Wq : (z == 1) ? Wk : Wv;
  const float* bias = (z == 0) ? bq : (z == 1) ? bk : bv;

  const int ra = tid >> 1, ca = (tid & 1) * 16;   // A tile: 2x16B per thread
  const int rb = tid >> 2, cb = (tid & 3) * 8;    // B tile: 1x16B per thread
  const bf16_t* gA = A + (size_t)(m0 + ra) * D_ + ca;
  const bf16_t* gB = W + (size_t)(n0 + rb) * D_ + cb;

  // prologue: fill buffer 0
  cp16(&As[0][ra * 40 + ca], gA);
  cp16(&As[0][ra * 40 + ca + 8], gA + 8);
  cp16(&Bs[0][rb * 40 + cb], gB);
  wait_async0();
  __syncthreads();

  v8f acc[2][2] = {{{}, {}}, {{}, {}}};
  for (int k0 = 0; k0 < D_; k0 += 32) {
    const int cur = (k0 >> 5) & 1;
    if (k0 + 32 < D_) {                     // prefetch next tile into other buffer
      const int nxt = cur ^ 1;
      cp16(&As[nxt][ra * 40 + ca],     gA + k0 + 32);
      cp16(&As[nxt][ra * 40 + ca + 8], gA + k0 + 40);
      cp16(&Bs[nxt][rb * 40 + cb],     gB + k0 + 32);
    }
    v16bf a0 = frag_a(&As[cur][(wm * 32) * 40], 40, lane);
    v16bf a1 = frag_a(&As[cur][(wm * 32 + 16) * 40], 40, lane);
    v16bf b0 = frag_b(&Bs[cur][(wn * 32) * 40], 40, lane);
    v16bf b1 = frag_b(&Bs[cur][(wn * 32 + 16) * 40], 40, lane);
    acc[0][0] = wmma_bf16(a0, b0, acc[0][0]);
    acc[0][1] = wmma_bf16(a0, b1, acc[0][1]);
    acc[1][0] = wmma_bf16(a1, b0, acc[1][0]);
    acc[1][1] = wmma_bf16(a1, b1, acc[1][1]);
    if (k0 + 32 < D_) {
      wait_async0();
      __syncthreads();
    }
  }
  const int mb = (lane & 16) ? 8 : 0;
  const int nlo = lane & 15;
#pragma unroll
  for (int i = 0; i < 2; ++i)
#pragma unroll
    for (int j = 0; j < 2; ++j)
#pragma unroll
      for (int r2 = 0; r2 < 8; ++r2) {
        int m = m0 + wm * 32 + i * 16 + mb + r2;
        int n = n0 + wn * 32 + j * 16 + nlo;
        float val = acc[i][j][r2] + bias[n];
        int bidx = m >> 11, s = m & (S_ - 1);
        int h = n >> 6, hd = n & 63;
        if (z == 2)
          Vo[(((size_t)bidx * H_ + h) * HD_ + hd) * S_ + s] = to_bf16(val);       // [b,h,hd,s]
        else {
          bf16_t* dst = (z == 0) ? Qo : Ko;
          dst[(((size_t)bidx * H_ + h) * S_ + s) * HD_ + hd] = to_bf16(val);      // [b,h,s,hd]
        }
      }
}

// ---------------- fused spiking attention ----------------
__global__ __launch_bounds__(256) void attn_kernel(
    const bf16_t* __restrict__ Q, const bf16_t* __restrict__ K,
    const bf16_t* __restrict__ Vt, bf16_t* __restrict__ Ctx,
    float* __restrict__ usage_part) {
  __shared__ __align__(16) bf16_t Qs[64 * 72];
  __shared__ __align__(16) bf16_t Ks[64 * 72];
  __shared__ __align__(16) bf16_t Vs[64 * 72];
  __shared__ __align__(16) bf16_t Ps[64 * 72];
  __shared__ float Ls[64 * 66];
  __shared__ float rowm[64], Zr[64], Ssp[64], red[256];

  const int tid = threadIdx.x, lane = tid & 31, wave = tid >> 5;
  const int wm = wave & 3, wn = wave >> 2;
  const int q0 = blockIdx.x * 64;
  const int bh = blockIdx.y;
  const size_t headQK = (size_t)bh * S_ * HD_;
  const size_t headV  = (size_t)bh * HD_ * S_;
  const int row = tid >> 2, part = tid & 3;
  const int lr_ = tid >> 2, lc_ = (tid & 3) * 16;   // tile-fill mapping

  // Q tile (resident for whole block)
  cp16(&Qs[lr_ * 72 + lc_],     Q + headQK + (size_t)(q0 + lr_) * HD_ + lc_);
  cp16(&Qs[lr_ * 72 + lc_ + 8], Q + headQK + (size_t)(q0 + lr_) * HD_ + lc_ + 8);
  if (tid < 64) { rowm[tid] = -3.0e38f; Zr[tid] = 0.f; Ssp[tid] = 0.f; }
  wait_async0();
  __syncthreads();

  // ---- pass 1: online softmax stats (rowmax, Z) ----
  for (int t0 = 0; t0 < S_; t0 += 64) {
    cp16(&Ks[lr_ * 72 + lc_],     K + headQK + (size_t)(t0 + lr_) * HD_ + lc_);
    cp16(&Ks[lr_ * 72 + lc_ + 8], K + headQK + (size_t)(t0 + lr_) * HD_ + lc_ + 8);
    wait_async0();
    __syncthreads();
    v16bf a0 = frag_a(&Qs[(wm * 16) * 72], 72, lane);
    v16bf a1 = frag_a(&Qs[(wm * 16) * 72 + 32], 72, lane);
#pragma unroll
    for (int j = 0; j < 2; ++j) {
      v16bf b0 = frag_b(&Ks[(wn * 32 + j * 16) * 72], 72, lane);
      v16bf b1 = frag_b(&Ks[(wn * 32 + j * 16) * 72 + 32], 72, lane);
      v8f c = {};
      c = wmma_bf16(a0, b0, c);
      c = wmma_bf16(a1, b1, c);
      int mbase = wm * 16 + ((lane & 16) ? 8 : 0);
      int nb = wn * 32 + j * 16 + (lane & 15);
#pragma unroll
      for (int r2 = 0; r2 < 8; ++r2) Ls[(mbase + r2) * 66 + nb] = c[r2] * 0.125f;
    }
    __syncthreads();
    const float* lr = &Ls[row * 66 + part * 16];
    float mx = lr[0];
#pragma unroll
    for (int i = 1; i < 16; ++i) mx = fmaxf(mx, lr[i]);
    red[tid] = mx;
    __syncthreads();
    if (part == 0) {
      float cm = fmaxf(fmaxf(red[tid], red[tid + 1]), fmaxf(red[tid + 2], red[tid + 3]));
      float mo = rowm[row], mn = fmaxf(mo, cm);
      Zr[row] *= __expf(mo - mn);
      rowm[row] = mn;
    }
    __syncthreads();
    float mn = rowm[row], s = 0.f;
#pragma unroll
    for (int i = 0; i < 16; ++i) s += __expf(lr[i] - mn);
    red[tid] = s;
    __syncthreads();
    if (part == 0) Zr[row] += red[tid] + red[tid + 1] + red[tid + 2] + red[tid + 3];
    __syncthreads();
  }

  // ---- pass 2: threshold + context accumulation ----
  v8f oc0 = {}, oc1 = {};
  for (int t0 = 0; t0 < S_; t0 += 64) {
    cp16(&Ks[lr_ * 72 + lc_],     K + headQK + (size_t)(t0 + lr_) * HD_ + lc_);
    cp16(&Ks[lr_ * 72 + lc_ + 8], K + headQK + (size_t)(t0 + lr_) * HD_ + lc_ + 8);
    cp16(&Vs[lr_ * 72 + lc_],     Vt + headV + (size_t)lr_ * S_ + t0 + lc_);
    cp16(&Vs[lr_ * 72 + lc_ + 8], Vt + headV + (size_t)lr_ * S_ + t0 + lc_ + 8);
    wait_async0();
    __syncthreads();
    v16bf a0 = frag_a(&Qs[(wm * 16) * 72], 72, lane);
    v16bf a1 = frag_a(&Qs[(wm * 16) * 72 + 32], 72, lane);
#pragma unroll
    for (int j = 0; j < 2; ++j) {
      v16bf b0 = frag_b(&Ks[(wn * 32 + j * 16) * 72], 72, lane);
      v16bf b1 = frag_b(&Ks[(wn * 32 + j * 16) * 72 + 32], 72, lane);
      v8f c = {};
      c = wmma_bf16(a0, b0, c);
      c = wmma_bf16(a1, b1, c);
      int mbase = wm * 16 + ((lane & 16) ? 8 : 0);
      int nb = wn * 32 + j * 16 + (lane & 15);
#pragma unroll
      for (int r2 = 0; r2 < 8; ++r2) Ls[(mbase + r2) * 66 + nb] = c[r2] * 0.125f;
    }
    __syncthreads();
    {
      const float* lr = &Ls[row * 66 + part * 16];
      float mn = rowm[row], thr = 0.1f * Zr[row], psum = 0.f;
#pragma unroll
      for (int i = 0; i < 16; ++i) {
        float e = __expf(lr[i] - mn);
        float sp = (e > thr) ? e : 0.f;     // probs>0.1 <=> e > 0.1*Z
        psum += sp;
        Ps[row * 72 + part * 16 + i] = to_bf16(sp);
      }
      red[tid] = psum;
    }
    __syncthreads();
    if (part == 0) Ssp[row] += red[tid] + red[tid + 1] + red[tid + 2] + red[tid + 3];
    __syncthreads();
    // ctx += P(64x64) @ V(64x64)
    v16bf pa0 = frag_a(&Ps[(wm * 16) * 72], 72, lane);
    v16bf pa1 = frag_a(&Ps[(wm * 16) * 72 + 32], 72, lane);
    v16bf vb0 = frag_b(&Vs[(wn * 32) * 72], 72, lane);
    v16bf vb1 = frag_b(&Vs[(wn * 32) * 72 + 32], 72, lane);
    oc0 = wmma_bf16(pa0, vb0, oc0);
    oc0 = wmma_bf16(pa1, vb1, oc0);
    v16bf vb2 = frag_b(&Vs[(wn * 32 + 16) * 72], 72, lane);
    v16bf vb3 = frag_b(&Vs[(wn * 32 + 16) * 72 + 32], 72, lane);
    oc1 = wmma_bf16(pa0, vb2, oc1);
    oc1 = wmma_bf16(pa1, vb3, oc1);
    __syncthreads();
  }

  // epilogue: attn = spiked/(sum_spiked + eps) => ctx * 1/(Ssp + eps*Z)
  const int b = bh >> 4, h = bh & 15;
  {
    int mbase = wm * 16 + ((lane & 16) ? 8 : 0);
    int nlo = lane & 15;
#pragma unroll
    for (int r2 = 0; r2 < 8; ++r2) {
      int m = mbase + r2;
      float inv = 1.0f / (Ssp[m] + 1e-8f * Zr[m]);
      size_t grow = (size_t)(b * S_ + q0 + m) * (H_ * HD_) + h * HD_ + wn * 32 + nlo;
      Ctx[grow]      = to_bf16(oc0[r2] * inv);
      Ctx[grow + 16] = to_bf16(oc1[r2] * inv);
    }
  }
  __syncthreads();
  if (tid < 64) red[tid] = Ssp[tid] / (Ssp[tid] + 1e-8f * Zr[tid]);
  __syncthreads();
  if (tid == 0) {
    float s = 0.f;
    for (int i = 0; i < 64; ++i) s += red[i];
    usage_part[bh * 32 + blockIdx.x] = s;   // deterministic partial (no atomics)
  }
}

// ---------------- output projection GEMM (fp32 out, double-buffered async) ----------------
__global__ __launch_bounds__(256) void gemm_out_kernel(
    const bf16_t* __restrict__ A, const bf16_t* __restrict__ W,
    const float* __restrict__ bo, float* __restrict__ AO) {
  __shared__ __align__(16) bf16_t As[2][128 * 40];
  __shared__ __align__(16) bf16_t Bs[2][64 * 40];
  const int tid = threadIdx.x, lane = tid & 31, wave = tid >> 5;
  const int wm = wave & 3, wn = wave >> 2;
  const int m0 = blockIdx.x * 128, n0 = blockIdx.y * 64;
  const int ra = tid >> 1, ca = (tid & 1) * 16;
  const int rb = tid >> 2, cb = (tid & 3) * 8;
  const bf16_t* gA = A + (size_t)(m0 + ra) * D_ + ca;
  const bf16_t* gB = W + (size_t)(n0 + rb) * D_ + cb;

  cp16(&As[0][ra * 40 + ca], gA);
  cp16(&As[0][ra * 40 + ca + 8], gA + 8);
  cp16(&Bs[0][rb * 40 + cb], gB);
  wait_async0();
  __syncthreads();

  v8f acc[2][2] = {{{}, {}}, {{}, {}}};
  for (int k0 = 0; k0 < D_; k0 += 32) {
    const int cur = (k0 >> 5) & 1;
    if (k0 + 32 < D_) {
      const int nxt = cur ^ 1;
      cp16(&As[nxt][ra * 40 + ca],     gA + k0 + 32);
      cp16(&As[nxt][ra * 40 + ca + 8], gA + k0 + 40);
      cp16(&Bs[nxt][rb * 40 + cb],     gB + k0 + 32);
    }
    v16bf a0 = frag_a(&As[cur][(wm * 32) * 40], 40, lane);
    v16bf a1 = frag_a(&As[cur][(wm * 32 + 16) * 40], 40, lane);
    v16bf b0 = frag_b(&Bs[cur][(wn * 32) * 40], 40, lane);
    v16bf b1 = frag_b(&Bs[cur][(wn * 32 + 16) * 40], 40, lane);
    acc[0][0] = wmma_bf16(a0, b0, acc[0][0]);
    acc[0][1] = wmma_bf16(a0, b1, acc[0][1]);
    acc[1][0] = wmma_bf16(a1, b0, acc[1][0]);
    acc[1][1] = wmma_bf16(a1, b1, acc[1][1]);
    if (k0 + 32 < D_) {
      wait_async0();
      __syncthreads();
    }
  }
  const int mb = (lane & 16) ? 8 : 0;
  const int nlo = lane & 15;
#pragma unroll
  for (int i = 0; i < 2; ++i)
#pragma unroll
    for (int j = 0; j < 2; ++j)
#pragma unroll
      for (int r2 = 0; r2 < 8; ++r2) {
        int m = m0 + wm * 32 + i * 16 + mb + r2;
        int n = n0 + wn * 32 + j * 16 + nlo;
        AO[(size_t)m * D_ + n] = acc[i][j][r2] + bo[n];
      }
}

// ---------------- layernorm ----------------
__global__ __launch_bounds__(256) void ln_kernel(const float* __restrict__ AO,
                                                 const float* __restrict__ scale,
                                                 const float* __restrict__ offset,
                                                 float* __restrict__ Y) {
  __shared__ float r1[256], r2[256];
  const int row = blockIdx.x, tid = threadIdx.x;
  const float* a = AO + (size_t)row * D_;
  float v[4], s = 0.f, sq = 0.f;
#pragma unroll
  for (int i = 0; i < 4; ++i) { v[i] = a[tid + i * 256]; s += v[i]; sq += v[i] * v[i]; }
  r1[tid] = s; r2[tid] = sq;
  __syncthreads();
  for (int off = 128; off > 0; off >>= 1) {
    if (tid < off) { r1[tid] += r1[tid + off]; r2[tid] += r2[tid + off]; }
    __syncthreads();
  }
  float mean = r1[0] * (1.0f / D_);
  float var = r2[0] * (1.0f / D_) - mean * mean;
  float rs = rsqrtf(var + 1e-5f);
  float* y = Y + (size_t)row * D_;
#pragma unroll
  for (int i = 0; i < 4; ++i) {
    int c = tid + i * 256;
    y[c] = (v[i] - mean) * rs * scale[c] + offset[c];
  }
}

__global__ void usage_kernel(const float* __restrict__ part, float* __restrict__ out) {
  int h = threadIdx.x;
  if (h < H_) {
    float s = 0.f;
    for (int b = 0; b < B_; ++b)
      for (int q = 0; q < 32; ++q) s += part[(b * H_ + h) * 32 + q];
    out[h] = s * (1.0f / ((float)B_ * (float)S_ * (float)S_));
  }
}

extern "C" void kernel_launch(void* const* d_in, const int* in_sizes, int n_in,
                              void* d_out, int out_size, void* d_ws, size_t ws_size,
                              hipStream_t stream) {
  (void)in_sizes; (void)n_in; (void)out_size; (void)ws_size;
  const float* x   = (const float*)d_in[0];
  const float* wq  = (const float*)d_in[1];
  const float* bq  = (const float*)d_in[2];
  const float* wk  = (const float*)d_in[3];
  const float* bk  = (const float*)d_in[4];
  const float* wv  = (const float*)d_in[5];
  const float* bv  = (const float*)d_in[6];
  const float* wo  = (const float*)d_in[7];
  const float* bo  = (const float*)d_in[8];
  const float* lns = (const float*)d_in[9];
  const float* lno = (const float*)d_in[10];
  float* out = (float*)d_out;

  char* ws = (char*)d_ws;
  size_t off = 0;
  auto alloc = [&](size_t bytes) { char* p = ws + off; off += (bytes + 255) & ~(size_t)255; return p; };
  const size_t NTOK = (size_t)B_ * S_;               // 4096
  bf16_t* Xbf  = (bf16_t*)alloc(NTOK * D_ * 2);
  bf16_t* Wqb  = (bf16_t*)alloc((size_t)D_ * D_ * 2);
  bf16_t* Wkb  = (bf16_t*)alloc((size_t)D_ * D_ * 2);
  bf16_t* Wvb  = (bf16_t*)alloc((size_t)D_ * D_ * 2);
  bf16_t* Wob  = (bf16_t*)alloc((size_t)D_ * D_ * 2);
  bf16_t* Qb   = (bf16_t*)alloc(NTOK * D_ * 2);
  bf16_t* Kb   = (bf16_t*)alloc(NTOK * D_ * 2);
  bf16_t* Vtb  = (bf16_t*)alloc(NTOK * D_ * 2);
  bf16_t* Ctx  = (bf16_t*)alloc(NTOK * D_ * 2);
  float*  AO   = (float*)alloc(NTOK * D_ * 4);
  float*  Upart= (float*)alloc((size_t)B_ * H_ * 32 * 4);

  const int nX = (int)(NTOK * D_);
  cvt_x_kernel<<<nX / 256, 256, 0, stream>>>(x, Xbf, nX);
  dim3 tg(32, 32);
  cvt_wT_kernel<<<tg, 256, 0, stream>>>(wq, Wqb);
  cvt_wT_kernel<<<tg, 256, 0, stream>>>(wk, Wkb);
  cvt_wT_kernel<<<tg, 256, 0, stream>>>(wv, Wvb);
  cvt_wT_kernel<<<tg, 256, 0, stream>>>(wo, Wob);

  gemm_qkv_kernel<<<dim3(32, 16, 3), 256, 0, stream>>>(Xbf, Wqb, Wkb, Wvb,
                                                       bq, bk, bv, Qb, Kb, Vtb);
  attn_kernel<<<dim3(32, 32), 256, 0, stream>>>(Qb, Kb, Vtb, Ctx, Upart);
  gemm_out_kernel<<<dim3(32, 16), 256, 0, stream>>>(Ctx, Wob, bo, AO);
  ln_kernel<<<(int)NTOK, 256, 0, stream>>>(AO, lns, lno, out);
  usage_kernel<<<1, 32, 0, stream>>>(Upart, out + NTOK * D_);
}
